// Multi_Loss_9809705304507
// MI455X (gfx1250) — matrix operand order repaired and checked by reading, not compiled, not run on hardware
//
#include <hip/hip_runtime.h>
#include <math.h>

// ---------------- problem constants (from reference setup_inputs) ------------
#define BB     8        // batch
#define PP     19248    // priors
#define CC     81       // classes
#define NOBJ   8
#define KDIM   32       // mask coef dim
#define PH     138
#define PW     138
#define NPIX   (PH*PW)  // 19044
#define HGT    550
#define WID    550
#define NSEL   112      // 100 masks-to-train rounded up to 7 WMMA N-tiles
#define NTILE  7
#define MASKS_TO_TRAIN 100
#define HBINS  65536

typedef __attribute__((ext_vector_type(16))) __bf16 v16bf;
typedef __attribute__((ext_vector_type(8)))  float  v8f;

// ---------------- K0: zero accumulator region --------------------------------
__global__ void k_zero(unsigned* __restrict__ p, int n) {
  int i = blockIdx.x * 256 + threadIdx.x;
  if (i < n) p[i] = 0u;
}

// ---------------- K1: prior matching + box loss ------------------------------
__global__ void k_match(const float* __restrict__ priors,
                        const float* __restrict__ box_gt,
                        const int*   __restrict__ class_gt,
                        const float* __restrict__ box_p,
                        float* __restrict__ epm, int* __restrict__ epi,
                        int* __restrict__ confgt,
                        int* __restrict__ num_pos, float* __restrict__ lacc) {
  const int b = blockIdx.x, t = threadIdx.x;
  __shared__ float sbg[NOBJ * 4];
  __shared__ unsigned long long skey[NOBJ];
  __shared__ float sredf[256];
  __shared__ int   sredi[256];
  if (t < NOBJ * 4) sbg[t] = box_gt[b * NOBJ * 4 + t];
  if (t < NOBJ) skey[t] = 0ull;
  __syncthreads();
  // phase A: per-prior best object + per-object best prior (packed-key atomicMax)
  for (int p = t; p < PP; p += 256) {
    const float* pr = priors + p * 4;
    float cx = pr[0], cy = pr[1], pw = pr[2], ph = pr[3];
    float dx1 = cx - 0.5f * pw, dy1 = cy - 0.5f * ph;
    float dx2 = cx + 0.5f * pw, dy2 = cy + 0.5f * ph;
    float areaD = pw * ph;
    float best = -1.f; int bj = 0;
#pragma unroll
    for (int j = 0; j < NOBJ; ++j) {
      float gx1 = sbg[4*j], gy1 = sbg[4*j+1], gx2 = sbg[4*j+2], gy2 = sbg[4*j+3];
      float iw = fminf(gx2, dx2) - fmaxf(gx1, dx1); iw = fmaxf(iw, 0.f);
      float ih = fminf(gy2, dy2) - fmaxf(gy1, dy1); ih = fmaxf(ih, 0.f);
      float inter = iw * ih;
      float iou = inter / ((gx2-gx1)*(gy2-gy1) + areaD - inter);
      if (iou > best) { best = iou; bj = j; }           // first-max tie-break
      unsigned long long key =
          ((unsigned long long)__float_as_uint(iou) << 32) |
          (unsigned)(0x7FFFFFFF - p);                   // smaller p wins ties
      atomicMax(&skey[j], key);
    }
    epm[b*PP + p] = best; epi[b*PP + p] = bj;
  }
  __syncthreads();
  // phase B: force each GT's best prior (serial -> last-wins like scatter set)
  if (t == 0) {
    for (int j = 0; j < NOBJ; ++j) {
      int q = 0x7FFFFFFF - (int)(unsigned)(skey[j] & 0xFFFFFFFFull);
      epm[b*PP + q] = 2.0f; epi[b*PP + q] = j;
    }
  }
  __syncthreads();
  // phase C: conf labels + fused smooth-L1 box loss over positives
  float lb = 0.f; int np = 0;
  for (int p = t; p < PP; p += 256) {
    float mx = epm[b*PP + p]; int j = epi[b*PP + p];
    int conf = class_gt[b*NOBJ + j] + 1;
    if (mx < 0.5f) conf = -1;
    if (mx < 0.4f) conf = 0;
    confgt[b*PP + p] = conf;
    if (conf > 0) {
      np++;
      const float* g  = &sbg[4*j];
      const float* pr = priors + p * 4;
      float o0 = (0.5f*(g[0]+g[2]) - pr[0]) / (0.1f * pr[2]);
      float o1 = (0.5f*(g[1]+g[3]) - pr[1]) / (0.1f * pr[3]);
      float o2 = __logf((g[2]-g[0]) / pr[2]) * 5.0f;    // 1/0.2
      float o3 = __logf((g[3]-g[1]) / pr[3]) * 5.0f;
      const float* bp = box_p + (size_t)(b*PP + p) * 4;
      float o[4] = {o0, o1, o2, o3};
#pragma unroll
      for (int i = 0; i < 4; ++i) {
        float d = fabsf(bp[i] - o[i]);
        lb += (d < 1.f) ? 0.5f*d*d : d - 0.5f;
      }
    }
  }
  sredf[t] = lb; sredi[t] = np; __syncthreads();
  for (int off = 128; off; off >>= 1) {
    if (t < off) { sredf[t] += sredf[t+off]; sredi[t] += sredi[t+off]; }
    __syncthreads();
  }
  if (t == 0) { atomicAdd(&lacc[0], sredf[0]); num_pos[b] = sredi[0]; }
}

// ---------------- K2: wave-per-prior logsumexp, mark, NLL, histogram ---------
__global__ void k_conf_stats(const float* __restrict__ class_p,
                             const int*   __restrict__ confgt,
                             float* __restrict__ mark, float* __restrict__ nll,
                             unsigned* __restrict__ hist) {
  int gw   = (blockIdx.x * blockDim.x + threadIdx.x) >> 5;   // one wave / prior
  int lane = threadIdx.x & 31;
  int b = gw / PP;
  const float* xp = class_p + (size_t)gw * CC;
  float x0 = xp[lane];
  float x1 = xp[lane + 32];
  float x2 = (lane < CC - 64) ? xp[lane + 64] : -1e30f;
  float m = fmaxf(fmaxf(x0, x1), x2);
#pragma unroll
  for (int o = 16; o; o >>= 1) m = fmaxf(m, __shfl_xor(m, o, 32));
  float s = __expf(x0 - m) + __expf(x1 - m) +
            ((lane < CC - 64) ? __expf(x2 - m) : 0.f);
#pragma unroll
  for (int o = 16; o; o >>= 1) s += __shfl_xor(s, o, 32);
  if (lane == 0) {
    float lse = m + __logf(s);
    int cg  = confgt[gw];
    int tgt = (cg > 0) ? cg : 0;
    nll[gw] = lse - xp[tgt];
    float mk = -1.f;                    // non-candidate marker
    if (cg == 0) {                      // candidate for hard-negative mining
      mk = fmaxf(lse - x0, 0.f);        // x0 (lane0) == class_p[...,0]
      unsigned bin = __float_as_uint(mk) >> 16;   // monotone for mk>=0
      atomicAdd(&hist[b * HBINS + bin], 1u);
    }
    mark[gw] = mk;
  }
}

// ---------------- K3: per-batch rank threshold from histogram ----------------
__global__ void k_threshold(const unsigned* __restrict__ hist,
                            const int* __restrict__ num_pos,
                            int* __restrict__ cutoff, int* __restrict__ remain,
                            int* __restrict__ taken, float* __restrict__ scaleb) {
  int b = threadIdx.x; if (b >= BB) return;
  int npos = num_pos[b];
  long long need = 3LL * npos; if (need > PP - 1) need = PP - 1;
  int cb = 0x7FFFFFFF, r = 0;
  if (need > 0) {
    long long acc = 0; cb = -1;
    for (int bin = HBINS - 1; bin >= 0; --bin) {
      unsigned c = hist[b * HBINS + bin];
      if (acc + c >= need) { cb = bin; r = (int)(need - acc); break; }
      acc += c;
    }
  }
  cutoff[b] = cb; remain[b] = r; taken[b] = 0;
  scaleb[b] = (npos > MASKS_TO_TRAIN) ? (float)npos / (float)MASKS_TO_TRAIN : 1.f;
}

// ---------------- K4: confidence loss (positives + mined negatives) ----------
__global__ void k_conf_loss(const int* __restrict__ confgt,
                            const float* __restrict__ mark,
                            const float* __restrict__ nll,
                            const int* __restrict__ cutoff,
                            const int* __restrict__ remain,
                            int* __restrict__ taken,
                            float* __restrict__ lacc) {
  __shared__ float sred[256];
  int i = blockIdx.x * 256 + threadIdx.x;
  float v = 0.f;
  if (i < BB * PP) {
    int b  = i / PP;
    int cg = confgt[i];
    bool sel = (cg > 0);
    if (!sel && cg == 0) {
      int cb  = cutoff[b];
      int bin = (int)(__float_as_uint(mark[i]) >> 16);
      if (bin > cb) sel = true;
      else if (bin == cb && atomicAdd(&taken[b], 1) < remain[b]) sel = true;
    }
    if (sel) v = nll[i];
  }
  sred[threadIdx.x] = v; __syncthreads();
  for (int off = 128; off; off >>= 1) {
    if (threadIdx.x < off) sred[threadIdx.x] += sred[threadIdx.x + off];
    __syncthreads();
  }
  if (threadIdx.x == 0) atomicAdd(&lacc[1], sred[0]);
}

// ---------------- K5: bilinear 550->138 downsample, packed bitmask -----------
__global__ void k_downsample(const float* __restrict__ mask_gt,
                             unsigned* __restrict__ dm) {
  int i = blockIdx.x * 256 + threadIdx.x;
  if (i >= BB * NPIX) return;
  int b = i / NPIX, pix = i - b * NPIX;
  int y = pix / PW, x = pix - y * PW;
  const float sc = (float)HGT / (float)PH;
  float sy = (y + 0.5f) * sc - 0.5f, sx = (x + 0.5f) * sc - 0.5f;
  int y0 = (int)floorf(sy), x0 = (int)floorf(sx);
  float fy = sy - y0, fx = sx - x0;
  int y1 = min(y0 + 1, HGT - 1), x1 = min(x0 + 1, WID - 1);
  y0 = max(y0, 0); x0 = max(x0, 0);
  unsigned bits = 0;
#pragma unroll
  for (int j = 0; j < NOBJ; ++j) {
    const float* mg = mask_gt + (size_t)(b * NOBJ + j) * HGT * WID;
    float v00 = mg[y0*WID + x0], v01 = mg[y0*WID + x1];
    float v10 = mg[y1*WID + x0], v11 = mg[y1*WID + x1];
    float v = (1.f-fy)*((1.f-fx)*v00 + fx*v01) + fy*((1.f-fx)*v10 + fx*v11);
    if (v > 0.5f) bits |= 1u << j;
  }
  dm[i] = bits;
}

// ---------------- K6: ordered compaction of first min(npos,100) positives ----
__global__ void k_compact(const int* __restrict__ confgt,
                          const int* __restrict__ epi,
                          const float* __restrict__ box_gt,
                          const float* __restrict__ coef_p,
                          const int* __restrict__ num_pos,
                          float* __restrict__ sel_coef, float* __restrict__ sel_crop,
                          float* __restrict__ sel_denom, float* __restrict__ sel_valid,
                          int* __restrict__ sel_pgi, float* __restrict__ maskacc) {
  const int b = blockIdx.x, t = threadIdx.x;
  __shared__ int s_base, s_wcnt[8], s_woff[8];
  if (t == 0) s_base = 0;
  if (t < NSEL) maskacc[b * NSEL + t] = 0.f;   // per-call accumulator init
  int nlimit = min(num_pos[b], MASKS_TO_TRAIN);
  __syncthreads();
  for (int start = 0; start < PP; start += 256) {
    int p = start + t;
    bool f = (p < PP) && (confgt[b*PP + p] > 0);
    unsigned m32 = (unsigned)__ballot(f);       // wave32
    int wid = t >> 5, lane = t & 31;
    if (lane == 0) s_wcnt[wid] = __popc(m32);
    __syncthreads();
    if (t == 0) {
      int acc = s_base;
      for (int w = 0; w < 8; ++w) { s_woff[w] = acc; acc += s_wcnt[w]; }
      s_base = acc;
    }
    __syncthreads();
    if (f) {
      int slot = s_woff[wid] + __popc(m32 & ((1u << lane) - 1u));
      if (slot < nlimit) {
        int j = epi[b*PP + p];
        const float* g = box_gt + (b * NOBJ + j) * 4;
        float xa = fminf(g[0], g[2]) * PW, xb = fmaxf(g[0], g[2]) * PW;
        float ya = fminf(g[1], g[3]) * PH, yb = fmaxf(g[1], g[3]) * PH;
        float* cr = sel_crop + (b * NSEL + slot) * 4;
        cr[0] = fmaxf(xa - 1.f, 0.f);  cr[1] = fminf(xb + 1.f, (float)PW);
        cr[2] = fmaxf(ya - 1.f, 0.f);  cr[3] = fminf(yb + 1.f, (float)PH);
        sel_denom[b*NSEL + slot] = (g[2]-g[0]) * (g[3]-g[1]);
        sel_valid[b*NSEL + slot] = 1.f;
        sel_pgi  [b*NSEL + slot] = j;
        const float* cf = coef_p + (size_t)(b*PP + p) * KDIM;
        float* dc = sel_coef + (size_t)(b*NSEL + slot) * KDIM;
        for (int k = 0; k < KDIM; ++k) dc[k] = cf[k];
      }
    }
    __syncthreads();
  }
  if (t < NSEL && t >= nlimit) {               // zero pad slots (valid=0)
    float* cr = sel_crop + (b*NSEL + t) * 4;
    cr[0] = cr[1] = cr[2] = cr[3] = 0.f;
    sel_denom[b*NSEL + t] = 1.f; sel_valid[b*NSEL + t] = 0.f;
    sel_pgi[b*NSEL + t] = 0;
    float* dc = sel_coef + (size_t)(b*NSEL + t) * KDIM;
    for (int k = 0; k < KDIM; ++k) dc[k] = 0.f;
  }
}

// ---------------- K7: WMMA mask loss (proto x coef^T, sigmoid/crop/BCE) ------
// BCE with the reference's clip(p,1e-7,1-1e-7) done division/log1p-free:
//   -log(sigmoid(z))   = softplus(-z)            (capped at -log(1e-7))
//   -log(1-sigmoid(z)) = softplus(-z) + z        (capped at -log(1e-7))
//   cropped-out pixel (cm=0): p clamps to 1e-7 -> contributes g * (-log 1e-7)
__global__ void __launch_bounds__(256) k_mask_wmma(
    const float* __restrict__ proto_p, const unsigned* __restrict__ dm,
    const float* __restrict__ sel_coef, const float* __restrict__ sel_crop,
    const int* __restrict__ sel_pgi, float* __restrict__ maskacc) {
  const int b = blockIdx.y, t = threadIdx.x;
  const float CL = 16.118095651f;              // -log(1e-7)
  __shared__ __bf16 s_coef[NSEL * KDIM];       // B-matrix rows, bf16
  __shared__ float  s_crop[NSEL * 4];
  __shared__ int    s_pgi[NSEL];
  __shared__ float  s_macc[NSEL];
  for (int i = t; i < NSEL * KDIM; i += 256)
    s_coef[i] = (__bf16)sel_coef[(size_t)b * NSEL * KDIM + i];
  for (int i = t; i < NSEL * 4; i += 256) s_crop[i] = sel_crop[b * NSEL * 4 + i];
  if (t < NSEL) { s_pgi[t] = sel_pgi[b * NSEL + t]; s_macc[t] = 0.f; }
  __syncthreads();

  const int wid = t >> 5, lane = t & 31;
  const int pixbase = (blockIdx.x * 8 + wid) * 16;   // 16 pixels per wave
  if (pixbase < NPIX) {                              // wave-uniform (EXEC all-1s)
    const int half = lane >> 4;                      // 0: lanes 0-15, 1: 16-31
    const int lrow = lane & 15;
    int arow = pixbase + lrow; if (arow > NPIX - 1) arow = NPIX - 1;
    const float* ap = proto_p + ((size_t)b * NPIX + arow) * KDIM;
    // A 16x32 bf16 layout: lanes0-15 K=0..15; lanes16-31 K=16..31
    v16bf av;
#pragma unroll
    for (int i = 0; i < 8; ++i) av[i]     = (__bf16)ap[half * 16 + i];
#pragma unroll
    for (int i = 0; i < 8; ++i) av[8 + i] = (__bf16)ap[half * 16 + 8 + i];
    // prefetch next 128-pixel chunk's proto rows (global_prefetch_b8)
    __builtin_prefetch(ap + 128 * KDIM, 0, 0);
    // per-pixel epilogue data (D layout: VGPR v -> M = v + 8*half)
    int pixv[8]; unsigned dmb[8]; float xco[8], yco[8];
#pragma unroll
    for (int v = 0; v < 8; ++v) {
      int pix = pixbase + half * 8 + v;
      pixv[v] = pix;
      dmb[v]  = (pix < NPIX) ? dm[b * NPIX + pix] : 0u;
      int yy  = pix / PW;
      yco[v] = (float)yy; xco[v] = (float)(pix - yy * PW);
    }
#pragma unroll
    for (int n = 0; n < NTILE; ++n) {
      int mrow = n * 16 + lrow;
      // B 32x16 bf16 layout: lanes0-15 K=0..15, lanes16-31 K=16..31 (N = lane&15)
      v16bf bv = *(const v16bf*)&s_coef[mrow * KDIM + half * 16];
      v8f acc = {0.f, 0.f, 0.f, 0.f, 0.f, 0.f, 0.f, 0.f};
      acc = __builtin_amdgcn_wmma_f32_16x16x32_bf16(
          /*neg_a=*/false, av, /*neg_b=*/false, bv,
          /*c_mod=*/(short)0, acc, /*reuse_a=*/false, /*reuse_b=*/false);
      float x1 = s_crop[mrow*4+0], x2 = s_crop[mrow*4+1];
      float y1 = s_crop[mrow*4+2], y2 = s_crop[mrow*4+3];
      int pg = s_pgi[mrow];
      float lsum = 0.f;
#pragma unroll
      for (int v = 0; v < 8; ++v) {
        float z   = acc[v];
        float inb = (pixv[v] < NPIX) ? 1.f : 0.f;
        float cm  = (xco[v] >= x1 && xco[v] < x2 &&
                     yco[v] >= y1 && yco[v] < y2) ? 1.f : 0.f;
        float g   = (float)((dmb[v] >> pg) & 1u);
        float spn = __logf(1.f + __expf(-z));        // softplus(-z)
        float bce = cm * (g * fminf(spn, CL) +
                          (1.f - g) * fminf(spn + z, CL))
                  + (1.f - cm) * g * CL;
        lsum += inb * bce;
      }
      lsum += __shfl_xor(lsum, 16, 32);     // combine M=0..7 with M=8..15
      if (half == 0) atomicAdd(&s_macc[mrow], lsum);
    }
  }
  __syncthreads();
  for (int i = t; i < NSEL; i += 256)
    if (s_macc[i] != 0.f) atomicAdd(&maskacc[b * NSEL + i], s_macc[i]);
}

// ---------------- K8: final combine ------------------------------------------
__global__ void k_final(const float* __restrict__ maskacc,
                        const float* __restrict__ sel_denom,
                        const float* __restrict__ sel_valid,
                        const float* __restrict__ scaleb,
                        const float* __restrict__ lacc,
                        float* __restrict__ out) {
  __shared__ float sred[256];
  int t = threadIdx.x;
  float lm = 0.f;
  for (int i = t; i < BB * NSEL; i += 256) {
    int b = i / NSEL;
    lm += maskacc[i] / sel_denom[i] * sel_valid[i] * scaleb[b];
  }
  sred[t] = lm; __syncthreads();
  for (int off = 128; off; off >>= 1) {
    if (t < off) sred[t] += sred[t + off];
    __syncthreads();
  }
  if (t == 0)
    out[0] = 1.5f * lacc[0]                       // BBOX_ALPHA
           + 1.0f * lacc[1]                       // CONF_ALPHA
           + sred[0] * 6.125f / ((float)PH * (float)PW); // MASK_ALPHA/ph/pw
}

// ---------------- host-side wiring -------------------------------------------
extern "C" void kernel_launch(void* const* d_in, const int* in_sizes, int n_in,
                              void* d_out, int out_size, void* d_ws, size_t ws_size,
                              hipStream_t stream) {
  const float* class_p = (const float*)d_in[0];
  const float* box_p   = (const float*)d_in[1];
  const float* coef_p  = (const float*)d_in[2];
  const float* proto_p = (const float*)d_in[3];
  const float* priors  = (const float*)d_in[4];
  const float* box_gt  = (const float*)d_in[5];
  const float* mask_gt = (const float*)d_in[6];
  const int*   class_gt= (const int*)  d_in[7];
  float* out = (float*)d_out;

  // workspace carve-up (u32 units), total ~5.9 MB
  unsigned* W = (unsigned*)d_ws;
  size_t off = 0;
  float*    epm      = (float*)(W + off); off += (size_t)BB * PP;
  int*      epi      = (int*)  (W + off); off += (size_t)BB * PP;
  int*      confgt   = (int*)  (W + off); off += (size_t)BB * PP;
  float*    markv    = (float*)(W + off); off += (size_t)BB * PP;
  float*    nllv     = (float*)(W + off); off += (size_t)BB * PP;
  unsigned* dmv      = (unsigned*)(W + off); off += (size_t)BB * NPIX;
  unsigned* hist     = (unsigned*)(W + off); size_t zero_base = off;
                                          off += (size_t)BB * HBINS;
  int*      num_pos  = (int*)  (W + off); off += BB;
  int*      takenv   = (int*)  (W + off); off += BB;
  int*      cutoffv  = (int*)  (W + off); off += BB;
  int*      remainv  = (int*)  (W + off); off += BB;
  float*    scaleb   = (float*)(W + off); off += BB;
  float*    lacc     = (float*)(W + off); off += 2;
  size_t zero_n = off - zero_base;        // hist..lacc re-zeroed every call
  float*    sel_coef = (float*)(W + off); off += (size_t)BB * NSEL * KDIM;
  float*    sel_crop = (float*)(W + off); off += (size_t)BB * NSEL * 4;
  float*    sel_den  = (float*)(W + off); off += (size_t)BB * NSEL;
  float*    sel_val  = (float*)(W + off); off += (size_t)BB * NSEL;
  int*      sel_pgi  = (int*)  (W + off); off += (size_t)BB * NSEL;
  float*    maskacc  = (float*)(W + off); off += (size_t)BB * NSEL;
  (void)in_sizes; (void)n_in; (void)out_size; (void)ws_size;

  k_zero<<<(int)((zero_n + 255) / 256), 256, 0, stream>>>(
      (unsigned*)(W + zero_base), (int)zero_n);

  k_match<<<BB, 256, 0, stream>>>(priors, box_gt, class_gt, box_p,
                                  epm, epi, confgt, num_pos, lacc);

  k_conf_stats<<<(BB * PP) / 8, 256, 0, stream>>>(class_p, confgt,
                                                  markv, nllv, hist);

  k_downsample<<<(BB * NPIX + 255) / 256, 256, 0, stream>>>(mask_gt, dmv);

  k_threshold<<<1, 32, 0, stream>>>(hist, num_pos, cutoffv, remainv,
                                    takenv, scaleb);

  k_conf_loss<<<(BB * PP + 255) / 256, 256, 0, stream>>>(
      confgt, markv, nllv, cutoffv, remainv, takenv, lacc);

  k_compact<<<BB, 256, 0, stream>>>(confgt, epi, box_gt, coef_p, num_pos,
                                    sel_coef, sel_crop, sel_den, sel_val,
                                    sel_pgi, maskacc);

  dim3 mg((NPIX + 127) / 128, BB);
  k_mask_wmma<<<mg, 256, 0, stream>>>(proto_p, dmv, sel_coef, sel_crop,
                                      sel_pgi, maskacc);

  k_final<<<1, 256, 0, stream>>>(maskacc, sel_den, sel_val, scaleb, lacc, out);
}
